// GaussianSelfAttention_2027224564346
// MI455X (gfx1250) — compile-verified, matrix operand order checked by reading, not run on hardware
//
#include <hip/hip_runtime.h>
#include <hip/hip_bf16.h>
#include <math.h>

// ---------------------------------------------------------------------------
// GaussianSelfAttention for MI455X (gfx1250, wave32)
//   B=16, P=4096, D=512, G=64  -> QKV gemms with v_wmma_f32_16x16x32_bf16
//   Round 3: rolled K-loop (no 512-VGPR hoist/spill) + software-pipelined
//   B-fragment prefetch. 4 WMMAs per B fetch, A re-read from LDS each step.
// ---------------------------------------------------------------------------

typedef __attribute__((ext_vector_type(16))) __bf16 v16bf;
typedef __attribute__((ext_vector_type(8)))  __bf16 v8bf;
typedef __attribute__((ext_vector_type(8)))  float  v8f;

#define B_    16
#define P_    4096
#define D_    512
#define G_    64
#define NTILE 96          // 3 matrices (q,k,v) * 32 column tiles of 16
#define KB    16          // 512 / 32 k-steps
#define MBLK  64          // rows per workgroup (4 M-subtiles of 16)
#define TOT   (B_ * P_ * D_)   // 33554432 elements per q/k/v matrix

// ---------------------------------------------------------------------------
// Pack Wq/Wk/Wv (fp32 row-major [n][k], computing x @ W^T) into bf16 fragments
// laid out exactly as the WMMA B-operand wants per lane:
//   lane l in [0,16): col n = l,  K = {0..7, 16..23} of the 32-k block
//   lane l in [16,32): col n=l-16, K = {8..15, 24..31}
// flat element offset = (((tt*16 + kb)*32 + lane) * 16)
// ---------------------------------------------------------------------------
__global__ void prepack_w_kernel(const float* __restrict__ Wq,
                                 const float* __restrict__ Wk,
                                 const float* __restrict__ Wv,
                                 __bf16* __restrict__ out) {
    int idx = blockIdx.x * blockDim.x + threadIdx.x;   // 96*16*32 = 49152
    if (idx >= NTILE * KB * 32) return;
    int lane = idx & 31;
    int kb   = (idx >> 5) & 15;
    int tt   = idx >> 9;                 // 0..95
    int mat  = tt >> 5;                  // 0=q,1=k,2=v
    int t    = tt & 31;
    const float* W = (mat == 0) ? Wq : ((mat == 1) ? Wk : Wv);
    int n     = t * 16 + (lane & 15);
    int kbase = kb * 32 + ((lane >> 4) << 3);
    __bf16* dst = out + (size_t)idx * 16;
#pragma unroll
    for (int j = 0; j < 8; j++) dst[j]     = (__bf16)W[(size_t)n * D_ + kbase + j];
#pragma unroll
    for (int j = 0; j < 8; j++) dst[8 + j] = (__bf16)W[(size_t)n * D_ + kbase + 16 + j];
}

// ---------------------------------------------------------------------------
// Fused QKV GEMM: C[m, mat*512 + n] = sum_k x[m,k] * W_mat[n,k] + b_mat[n]
// One workgroup (8 wave32) per 64-row M block. A panel (64x512 bf16, 64KB)
// staged in LDS. Each wave owns 12 of the 96 n-tiles with 4 accumulators
// (one per 16-row M-subtile): each B fragment feeds 4 WMMAs. The K loop is
// kept ROLLED (unroll 1) so A fragments are streamed from LDS per step
// instead of being hoisted into (spilled) registers; the next B fragment is
// prefetched ahead of the current step's WMMAs.
// ---------------------------------------------------------------------------
__global__ __launch_bounds__(256) void qkv_gemm_kernel(
        const float*  __restrict__ x,
        const __bf16* __restrict__ wpack,
        const float*  __restrict__ bq,
        const float*  __restrict__ bk,
        const float*  __restrict__ bv,
        __bf16*       __restrict__ qkv) {
    __shared__ __align__(16) __bf16 As[MBLK * D_];   // 64 KB of 320 KB WGP LDS

    const int tid = threadIdx.x;
    const size_t m0 = (size_t)blockIdx.x * MBLK;

    // ---- cooperative stage: 64 x 512 fp32 -> bf16 in LDS (row-major) ----
    {
        int r  = tid >> 2;               // 0..63
        int c0 = (tid & 3) << 7;         // 0,128,256,384
        const float4* src = (const float4*)(x + (m0 + r) * D_ + c0);
#pragma unroll
        for (int i = 0; i < 32; i++) {
            float4 f = src[i];
            int o = r * D_ + c0 + i * 4;
            As[o + 0] = (__bf16)f.x;
            As[o + 1] = (__bf16)f.y;
            As[o + 2] = (__bf16)f.z;
            As[o + 3] = (__bf16)f.w;
        }
    }
    __syncthreads();

    const int w    = tid >> 5;
    const int lane = tid & 31;
    const int rA   = lane & 15;             // A row (within subtile) this lane carries
    const int ksel = (lane >> 4) << 3;      // +8 k shift for upper half-wave

    for (int tt = w; tt < NTILE; tt += 8) {
        v8f acc0 = {}, acc1 = {}, acc2 = {}, acc3 = {};
        const __bf16* bp = wpack + ((size_t)tt * KB * 32 + lane) * 16;
        // prime the B pipeline (kb = 0)
        v16bf bfrag = *(const v16bf*)(bp);
#pragma unroll 1
        for (int kb = 0; kb < KB; kb++) {
            // prefetch next B fragment (wraps harmlessly on the last step)
            int kn = (kb + 1) & (KB - 1);
            v16bf bnext = *(const v16bf*)(bp + (size_t)kn * 32 * 16);

            int kbase = kb * 32 + ksel;
            // A fragments for the 4 M-subtiles: two ds_load_b128 each
            union { v16bf v; v8bf h[2]; } a0, a1, a2, a3;
            a0.h[0] = *(const v8bf*)&As[(rA +  0) * D_ + kbase];
            a0.h[1] = *(const v8bf*)&As[(rA +  0) * D_ + kbase + 16];
            a1.h[0] = *(const v8bf*)&As[(rA + 16) * D_ + kbase];
            a1.h[1] = *(const v8bf*)&As[(rA + 16) * D_ + kbase + 16];
            a2.h[0] = *(const v8bf*)&As[(rA + 32) * D_ + kbase];
            a2.h[1] = *(const v8bf*)&As[(rA + 32) * D_ + kbase + 16];
            a3.h[0] = *(const v8bf*)&As[(rA + 48) * D_ + kbase];
            a3.h[1] = *(const v8bf*)&As[(rA + 48) * D_ + kbase + 16];

            acc0 = __builtin_amdgcn_wmma_f32_16x16x32_bf16(
                    false, a0.v, false, bfrag, (short)0, acc0, false, false);
            acc1 = __builtin_amdgcn_wmma_f32_16x16x32_bf16(
                    false, a1.v, false, bfrag, (short)0, acc1, false, false);
            acc2 = __builtin_amdgcn_wmma_f32_16x16x32_bf16(
                    false, a2.v, false, bfrag, (short)0, acc2, false, false);
            acc3 = __builtin_amdgcn_wmma_f32_16x16x32_bf16(
                    false, a3.v, false, bfrag, (short)0, acc3, false, false);

            bfrag = bnext;
        }
        // ---- epilogue: C layout -> VGPR r, lanes 0-15: M=r, lanes 16-31: M=r+8
        int mat  = tt >> 5;
        int n0   = (tt & 31) << 4;
        int col  = n0 + (lane & 15);
        int half = lane >> 4;
        const float* biasp = (mat == 0) ? bq : ((mat == 1) ? bk : bv);
        float bias = biasp[col];
        __bf16* outp = qkv + (size_t)mat * TOT + (m0 + half * 8) * D_ + col;
#pragma unroll
        for (int r = 0; r < 8; r++)
            outp[(size_t)(r +  0) * D_] = (__bf16)(acc0[r] + bias);
#pragma unroll
        for (int r = 0; r < 8; r++)
            outp[(size_t)(r + 16) * D_] = (__bf16)(acc1[r] + bias);
#pragma unroll
        for (int r = 0; r < 8; r++)
            outp[(size_t)(r + 32) * D_] = (__bf16)(acc2[r] + bias);
#pragma unroll
        for (int r = 0; r < 8; r++)
            outp[(size_t)(r + 48) * D_] = (__bf16)(acc3[r] + bias);
    }
}

// ---------------------------------------------------------------------------
// Grid coordinates: sample_{x,y} = tanh((noise + avg[img]) * std[img])
// Stored as concat([sample_x, sample_y]) per batch; the reference's
// (B,2,P)->(B,64,64,2) reshape means point p reads flat elems 2p and 2p+1.
// ---------------------------------------------------------------------------
__global__ void grid_kernel(const float* __restrict__ avgs,
                            const float* __restrict__ stds,
                            const float* __restrict__ nx,
                            const float* __restrict__ ny,
                            const int*   __restrict__ img_ids,
                            float*       __restrict__ gridbuf) {
    int idx = blockIdx.x * blockDim.x + threadIdx.x;
    if (idx >= B_ * P_) return;
    int b = idx >> 12, p = idx & (P_ - 1);
    int img = img_ids[b];
    const float* ab = avgs + (size_t)img * 2 * P_;
    const float* sb = stds + (size_t)img * 2 * P_;
    gridbuf[(size_t)b * 2 * P_ + p]      = tanhf((nx[idx] + ab[p])      * sb[p]);
    gridbuf[(size_t)b * 2 * P_ + P_ + p] = tanhf((ny[idx] + ab[P_ + p]) * sb[P_ + p]);
}

// ---------------------------------------------------------------------------
// Bilinear gather of k,v + score dot with q + sigmoid * v.
// One wave32 per output point; lane owns 16 channels (32B v16bf loads).
// 4-corner rows are L2-resident (k/v per batch ~ 8 MB bf16 vs 192 MB L2).
// ---------------------------------------------------------------------------
__global__ __launch_bounds__(256) void attn_kernel(
        const __bf16* __restrict__ qkv,
        const float*  __restrict__ gridbuf,
        float*        __restrict__ out) {
    int w = threadIdx.x >> 5, lane = threadIdx.x & 31;
    int pp = blockIdx.x * 8 + w;                 // 0 .. 65535
    int b = pp >> 12, p = pp & (P_ - 1);

    const __bf16* qb  = qkv;
    const __bf16* kbm = qkv + (size_t)TOT;
    const __bf16* vbm = qkv + (size_t)2 * TOT;

    float gx = gridbuf[(size_t)b * 2 * P_ + 2 * p];
    float gy = gridbuf[(size_t)b * 2 * P_ + 2 * p + 1];
    // align_corners=False mapping
    float ix = ((gx + 1.f) * (float)G_ - 1.f) * 0.5f;
    float iy = ((gy + 1.f) * (float)G_ - 1.f) * 0.5f;
    float fx0 = floorf(ix), fy0 = floorf(iy);
    int   ix0 = (int)fx0,   iy0 = (int)fy0;
    float wx1 = ix - fx0, wy1 = iy - fy0;
    float wx0 = 1.f - wx1, wy0 = 1.f - wy1;

    int   cx[4] = { ix0, ix0 + 1, ix0,     ix0 + 1 };
    int   cy[4] = { iy0, iy0,     iy0 + 1, iy0 + 1 };
    float cw[4] = { wx0 * wy0, wx1 * wy0, wx0 * wy1, wx1 * wy1 };

    int c0 = lane * 16;
    float sk[16], sv[16];
#pragma unroll
    for (int j = 0; j < 16; j++) { sk[j] = 0.f; sv[j] = 0.f; }

#pragma unroll
    for (int c = 0; c < 4; c++) {
        if (cx[c] < 0 || cx[c] >= G_ || cy[c] < 0 || cy[c] >= G_) continue;
        size_t pr = (size_t)(b * P_ + cy[c] * G_ + cx[c]) * D_ + c0;
        v16bf kv = *(const v16bf*)(kbm + pr);
        v16bf vv = *(const v16bf*)(vbm + pr);
        float wgt = cw[c];
#pragma unroll
        for (int j = 0; j < 16; j++) {
            sk[j] += wgt * (float)kv[j];
            sv[j] += wgt * (float)vv[j];
        }
    }

    v16bf qv = *(const v16bf*)(qb + (size_t)(b * P_ + p) * D_ + c0);
    float partial = 0.f;
#pragma unroll
    for (int j = 0; j < 16; j++) partial += (float)qv[j] * sk[j];
    // wave32 cross-lane reduction
#pragma unroll
    for (int off = 16; off >= 1; off >>= 1)
        partial += __shfl_xor(partial, off, 32);

    float sig = 1.f / (1.f + expf(-0.01f * partial));

    float* op = out + (size_t)(b * P_ + p) * D_ + c0;
#pragma unroll
    for (int j = 0; j < 16; j++) op[j] = sig * sv[j];
}

// ---------------------------------------------------------------------------
extern "C" void kernel_launch(void* const* d_in, const int* in_sizes, int n_in,
                              void* d_out, int out_size, void* d_ws, size_t ws_size,
                              hipStream_t stream) {
    const float* x    = (const float*)d_in[0];
    const float* Wq   = (const float*)d_in[1];
    const float* bq   = (const float*)d_in[2];
    const float* Wk   = (const float*)d_in[3];
    const float* bk   = (const float*)d_in[4];
    const float* Wv   = (const float*)d_in[5];
    const float* bv   = (const float*)d_in[6];
    const float* avgs = (const float*)d_in[7];
    const float* stds = (const float*)d_in[8];
    const float* nx   = (const float*)d_in[9];
    const float* ny   = (const float*)d_in[10];
    const int*   ids  = (const int*)d_in[11];
    float* out = (float*)d_out;

    // workspace layout:
    //   [0, 1.5MB)         packed bf16 weights
    //   [2MB, 2MB+201.3MB) q,k,v bf16 (3 * 33554432 elems)
    //   [+202MB, +512KB)   grid sample buffer (fp32)
    char* ws = (char*)d_ws;
    __bf16* wpack   = (__bf16*)ws;
    __bf16* qkv     = (__bf16*)(ws + ((size_t)2 << 20));
    float*  gridbuf = (float*) (ws + ((size_t)2 << 20) + ((size_t)202 << 20));

    prepack_w_kernel<<<192, 256, 0, stream>>>(Wq, Wk, Wv, wpack);
    qkv_gemm_kernel<<<(B_ * P_) / MBLK, 256, 0, stream>>>(x, wpack, bq, bk, bv, qkv);
    grid_kernel<<<(B_ * P_ + 255) / 256, 256, 0, stream>>>(avgs, stds, nx, ny, ids, gridbuf);
    attn_kernel<<<(B_ * P_) / 8, 256, 0, stream>>>(qkv, gridbuf, out);
}
